// TensorFusion_3994319585833
// MI455X (gfx1250) — matrix operand order repaired and checked by reading, not compiled, not run on hardware
//
#include <hip/hip_runtime.h>

typedef __attribute__((ext_vector_type(16))) _Float16 v16h;
typedef __attribute__((ext_vector_type(8)))  _Float16 v8h;
typedef __attribute__((ext_vector_type(8)))  float    v8f;
typedef __attribute__((ext_vector_type(4)))  int      v4i;

#define BATCH 256
#define DIM   31          // per-mode feature dim (bias col appended -> 32)
#define K1    32768       // (D+1)^3
#define H1    1024
#define H2    256
#define NCLS  4
#define KSPLIT 8          // split-K factor for GEMM1
#define KCHUNK (K1 / KSPLIT)   // 4096
#define NIT1   (KCHUNK / 32)   // 128 K-iterations per block in GEMM1

// ---- CDNA5 async global->LDS copy (ASYNCcnt-tracked), with safe fallback ----
#if defined(__gfx1250__) && __has_builtin(__builtin_amdgcn_global_load_async_to_lds_b128)
#define HAS_ASYNC_LDS 1
#else
#define HAS_ASYNC_LDS 0
#endif

static __device__ inline void async_copy_b128(const float* g, float* l) {
#if HAS_ASYNC_LDS
  // builtin signature (from clang diagnostic): arg0 is AS1 v4i*, non-const
  __builtin_amdgcn_global_load_async_to_lds_b128(
      (__attribute__((address_space(1))) v4i*)(void*)(float*)g,
      (__attribute__((address_space(3))) v4i*)(void*)l,
      /*offset=*/0, /*cpol=*/0);
#else
  *(float4*)l = *(const float4*)g;
#endif
}

// Wait until at most 2 async ops remain outstanding for this wave
// (the 2 b128 copies of the in-flight "next" tile).
static __device__ inline void wait_async_le2() {
#if HAS_ASYNC_LDS
#if __has_builtin(__builtin_amdgcn_s_wait_asynccnt)
  __builtin_amdgcn_s_wait_asynccnt(2);
#else
  asm volatile("s_wait_asynccnt 0x2" ::: "memory");
#endif
#endif
}

static __device__ inline v8f wmma_f16(v16h a, v16h b, v8f c) {
  // D = A(16x32 f16) * B(32x16 f16) + C(16x16 f32)
  return __builtin_amdgcn_wmma_f32_16x16x32_f16(
      /*neg_a=*/false, a, /*neg_b=*/false, b,
      /*c_mod=*/(short)0, c, /*reuse_a=*/false, /*reuse_b=*/false);
}

// -------------------------------------------------------------------------
// GEMM1: on-the-fly fused[256, 32768] (f16) @ W1[32768, 1024] (fp32->f16)
// grid = (N tiles = 16, M tiles = 2, K split = 8), block = 256 (8 waves)
// Block: 128(M) x 64(N) tile; wave grid 4x2; wave tile 32x32 (2x2 WMMA).
// W1 tiles are double-buffered in LDS via async global->LDS copies so the
// next tile's HBM fetch overlaps the current tile's WMMAs.
// Partial sums (fp32) per K-slice: h1part[blockIdx.z][256][1024].
// -------------------------------------------------------------------------
__launch_bounds__(256)
__global__ void k_gemm1(const float* __restrict__ x0,
                        const float* __restrict__ x1,
                        const float* __restrict__ x2,
                        const float* __restrict__ W1,
                        float* __restrict__ h1part)
{
  // z rows padded to stride 33 (odd) -> conflict-free strided reads
  __shared__ float sZ0[128 * 33];
  __shared__ float sZ1[128 * 33];
  // Shared region: z2 (setup only, 128*33*4 = 16896 B) overlapped with the
  // double-buffered W1 tile (2 * 32 * 68 * 4 = 17408 B).
  __shared__ __align__(16) char sScratch[2 * 32 * 68 * 4];
  float* const sZ2 = (float*)sScratch;         // valid during setup
  float* const sW  = (float*)sScratch;         // [2][32*68] during main loop

  const int tid  = threadIdx.x;
  const int lane = tid & 31;
  const int wave = tid >> 5;
  const int wm   = wave & 3;   // wave M index 0..3
  const int wn   = wave >> 2;  // wave N index 0..1

  const int mbase = blockIdx.y * 128;
  const int nbase = blockIdx.x * 64;
  const int kbase = blockIdx.z * KCHUNK;

  // ---- stage z0/z1/z2 (with bias column == 1.0) into LDS ----
  for (int idx = tid; idx < 128 * 32; idx += 256) {
    const int r = idx >> 5, c = idx & 31;
    const int gm = mbase + r;
    sZ0[r * 33 + c] = (c == DIM) ? 1.0f : x0[gm * DIM + c];
    sZ1[r * 33 + c] = (c == DIM) ? 1.0f : x1[gm * DIM + c];
    sZ2[r * 33 + c] = (c == DIM) ? 1.0f : x2[gm * DIM + c];
  }
  __syncthreads();

  const int rlo   = wm * 32 + (lane & 15);  // row within 128-tile (lo frag)
  const int rhi   = rlo + 16;               // hi frag
  const int klgrp = (lane >= 16) ? 8 : 0;   // A-frag K lane-group offset

  // cache z2 per lane in WMMA A-matrix K order (constant over all K chunks)
  float a2lo[16], a2hi[16];
#pragma unroll
  for (int t = 0; t < 16; ++t) {
    const int K = t + ((t >= 8) ? 8 : 0) + klgrp;  // 16x32 f16 A layout
    a2lo[t] = sZ2[rlo * 33 + K];
    a2hi[t] = sZ2[rhi * 33 + K];
  }
  __syncthreads();   // all z2 reads done; scratch region now owned by sW

  v8f acc[2][2];
#pragma unroll
  for (int mi = 0; mi < 2; ++mi)
#pragma unroll
    for (int ni = 0; ni < 2; ++ni)
      acc[mi][ni] = {};

  const int kr  = tid >> 4;         // 0..15   (tile K row for staging)
  const int nc4 = (tid & 15) << 2;  // 0..60   (tile N col, float4 granules)
  const int n0  = wn * 32 + (lane & 15);

  // prologue: kick off async copy of tile 0 into buffer 0
  async_copy_b128(W1 + (size_t)(kbase + kr)      * H1 + nbase + nc4,
                  sW + 0 * 2176 + kr * 68 + nc4);
  async_copy_b128(W1 + (size_t)(kbase + kr + 16) * H1 + nbase + nc4,
                  sW + 0 * 2176 + (kr + 16) * 68 + nc4);

  for (int it = 0; it < NIT1; ++it) {
    const int kk  = kbase + it * 32;
    const int cur = it & 1;
    const int nxt = cur ^ 1;

    // issue async copy of the NEXT tile (clamped dummy on last iteration so
    // the wait threshold stays uniform); previous compute on buffer `nxt`
    // finished before the barrier at the end of the previous iteration.
    const int knxt = (it + 1 < NIT1) ? (kk + 32) : kbase;
    async_copy_b128(W1 + (size_t)(knxt + kr)      * H1 + nbase + nc4,
                    sW + nxt * 2176 + kr * 68 + nc4);
    async_copy_b128(W1 + (size_t)(knxt + kr + 16) * H1 + nbase + nc4,
                    sW + nxt * 2176 + (kr + 16) * 68 + nc4);

    // deep L2 prefetch (~8 tiles ahead, clamped) -> global_prefetch_b8
    {
      int kpre = kk + 256;
      if (kpre > kbase + KCHUNK - 32) kpre = kbase + KCHUNK - 32;
      __builtin_prefetch((const void*)(W1 + (size_t)(kpre + kr) * H1 + nbase + nc4), 0, 1);
    }

    wait_async_le2();   // this wave's 2 ops for tile `cur` have landed in LDS
    __syncthreads();    // ...and every other wave's portion too

    // ---- build A fragments on the fly: fused = z0[i] * z1[j] * z2[k] ----
    const int i = (kk >> 10) & 31;
    const int j = (kk >> 5) & 31;
    const float zzlo = sZ0[rlo * 33 + i] * sZ1[rlo * 33 + j];
    const float zzhi = sZ0[rhi * 33 + i] * sZ1[rhi * 33 + j];
    v16h alo, ahi;
#pragma unroll
    for (int t = 0; t < 16; ++t) {
      alo[t] = (_Float16)(zzlo * a2lo[t]);
      ahi[t] = (_Float16)(zzhi * a2hi[t]);
    }

    // ---- B fragments from LDS (convert fp32 -> f16) ----
    const float* w = sW + cur * 2176;
    v16h b0, b1;
#pragma unroll
    for (int t = 0; t < 16; ++t) {
      const int K = t + ((t >= 8) ? 8 : 0) + ((lane >= 16) ? 8 : 0);  // 32x16 f16 B layout
      b0[t] = (_Float16)w[K * 68 + n0];
      b1[t] = (_Float16)w[K * 68 + n0 + 16];
    }

    acc[0][0] = wmma_f16(alo, b0, acc[0][0]);
    acc[0][1] = wmma_f16(alo, b1, acc[0][1]);
    acc[1][0] = wmma_f16(ahi, b0, acc[1][0]);
    acc[1][1] = wmma_f16(ahi, b1, acc[1][1]);

    __syncthreads();    // compute on `cur` done; next iter may overwrite it
  }

  // ---- write fp32 partials for this K slice (deterministic, no atomics) ----
  float* outp = h1part + (size_t)blockIdx.z * (BATCH * H1);
#pragma unroll
  for (int mi = 0; mi < 2; ++mi) {
#pragma unroll
    for (int ni = 0; ni < 2; ++ni) {
#pragma unroll
      for (int r = 0; r < 8; ++r) {
        const int row = mbase + wm * 32 + mi * 16 + r + ((lane >= 16) ? 8 : 0);
        const int col = nbase + wn * 32 + ni * 16 + (lane & 15);
        outp[(size_t)row * H1 + col] = acc[mi][ni][r];
      }
    }
  }
}

// -------------------------------------------------------------------------
// Reduce split-K partials + bias + ReLU + convert to f16 for GEMM2.
// -------------------------------------------------------------------------
__global__ void k_bias_relu1(const float* __restrict__ h1part,
                             const float* __restrict__ b1,
                             _Float16* __restrict__ h1h)
{
  const int i = blockIdx.x * blockDim.x + threadIdx.x;
  if (i < BATCH * H1) {
    float v = b1[i & (H1 - 1)];
#pragma unroll
    for (int s = 0; s < KSPLIT; ++s) v += h1part[(size_t)s * (BATCH * H1) + i];
    v = v > 0.0f ? v : 0.0f;
    h1h[i] = (_Float16)v;
  }
}

// -------------------------------------------------------------------------
// GEMM2: h1[256,1024] (f16) @ W2[1024,256] (fp32->f16) + b2, ReLU -> f16
// grid = (N tiles = 4, M tiles = 2), block = 256. Same wave tiling as GEMM1.
// -------------------------------------------------------------------------
__launch_bounds__(256)
__global__ void k_gemm2(const _Float16* __restrict__ A,
                        const float* __restrict__ W2,
                        const float* __restrict__ b2,
                        _Float16* __restrict__ h2h)
{
  __shared__ float sW[32 * 68];

  const int tid  = threadIdx.x;
  const int lane = tid & 31;
  const int wave = tid >> 5;
  const int wm   = wave & 3;
  const int wn   = wave >> 2;

  const int mbase = blockIdx.y * 128;
  const int nbase = blockIdx.x * 64;

  const int mlo   = mbase + wm * 32 + (lane & 15);
  const int mhi   = mlo + 16;
  const int klgrp = (lane >= 16) ? 8 : 0;
  const int kr    = tid >> 4;
  const int nc4   = (tid & 15) << 2;
  const int n0    = wn * 32 + (lane & 15);

  v8f acc[2][2];
#pragma unroll
  for (int mi = 0; mi < 2; ++mi)
#pragma unroll
    for (int ni = 0; ni < 2; ++ni)
      acc[mi][ni] = {};

  for (int it = 0; it < H1 / 32; ++it) {
    const int kk = it * 32;

    __syncthreads();
    const float4 w0 = *(const float4*)(W2 + (size_t)(kk + kr)      * H2 + nbase + nc4);
    const float4 w1 = *(const float4*)(W2 + (size_t)(kk + kr + 16) * H2 + nbase + nc4);
    *(float4*)(sW + kr * 68 + nc4)        = w0;
    *(float4*)(sW + (kr + 16) * 68 + nc4) = w1;
    __syncthreads();

    // A fragments: two aligned 8-half (16B) segments per frag
    const v8h a0lo = *(const v8h*)(A + (size_t)mlo * H1 + kk + klgrp);
    const v8h a1lo = *(const v8h*)(A + (size_t)mlo * H1 + kk + klgrp + 16);
    const v8h a0hi = *(const v8h*)(A + (size_t)mhi * H1 + kk + klgrp);
    const v8h a1hi = *(const v8h*)(A + (size_t)mhi * H1 + kk + klgrp + 16);
    const v16h alo = __builtin_shufflevector(a0lo, a1lo,
        0,1,2,3,4,5,6,7,8,9,10,11,12,13,14,15);
    const v16h ahi = __builtin_shufflevector(a0hi, a1hi,
        0,1,2,3,4,5,6,7,8,9,10,11,12,13,14,15);

    v16h b0, b1;
#pragma unroll
    for (int t = 0; t < 16; ++t) {
      const int K = t + ((t >= 8) ? 8 : 0) + ((lane >= 16) ? 8 : 0);
      b0[t] = (_Float16)sW[K * 68 + n0];
      b1[t] = (_Float16)sW[K * 68 + n0 + 16];
    }

    acc[0][0] = wmma_f16(alo, b0, acc[0][0]);
    acc[0][1] = wmma_f16(alo, b1, acc[0][1]);
    acc[1][0] = wmma_f16(ahi, b0, acc[1][0]);
    acc[1][1] = wmma_f16(ahi, b1, acc[1][1]);
  }

#pragma unroll
  for (int mi = 0; mi < 2; ++mi) {
#pragma unroll
    for (int ni = 0; ni < 2; ++ni) {
#pragma unroll
      for (int r = 0; r < 8; ++r) {
        const int row = mbase + wm * 32 + mi * 16 + r + ((lane >= 16) ? 8 : 0);
        const int col = nbase + wn * 32 + ni * 16 + (lane & 15);
        float v = acc[mi][ni][r] + b2[col];
        v = v > 0.0f ? v : 0.0f;
        h2h[(size_t)row * H2 + col] = (_Float16)v;
      }
    }
  }
}

// -------------------------------------------------------------------------
// Head: h2[256,256] (f16) @ W3[256,4] + b3 -> out[256,4] fp32 (scalar dot)
// -------------------------------------------------------------------------
__global__ void k_head(const _Float16* __restrict__ h2,
                       const float* __restrict__ W3,
                       const float* __restrict__ b3,
                       float* __restrict__ out)
{
  const int t = blockIdx.x * blockDim.x + threadIdx.x;
  if (t < BATCH * NCLS) {
    const int row = t >> 2, col = t & 3;
    float s = b3[col];
    for (int k = 0; k < H2; ++k)
      s += (float)h2[(size_t)row * H2 + k] * W3[k * NCLS + col];
    out[t] = s;
  }
}

extern "C" void kernel_launch(void* const* d_in, const int* in_sizes, int n_in,
                              void* d_out, int out_size, void* d_ws, size_t ws_size,
                              hipStream_t stream) {
  const float* x0 = (const float*)d_in[0];
  const float* x1 = (const float*)d_in[1];
  const float* x2 = (const float*)d_in[2];
  const float* W1 = (const float*)d_in[3];
  const float* b1 = (const float*)d_in[4];
  const float* W2 = (const float*)d_in[5];
  const float* b2 = (const float*)d_in[6];
  const float* W3 = (const float*)d_in[7];
  const float* b3 = (const float*)d_in[8];
  float* out = (float*)d_out;

  // workspace layout (~8.6 MB total)
  char* ws = (char*)d_ws;
  float*    h1part = (float*)ws;                                         // 8 MB
  _Float16* h1h    = (_Float16*)(ws + (size_t)KSPLIT * BATCH * H1 * 4);  // 512 KB
  _Float16* h2h    = (_Float16*)(ws + (size_t)KSPLIT * BATCH * H1 * 4
                                    + (size_t)BATCH * H1 * 2);           // 128 KB

  k_gemm1<<<dim3(H1 / 64, BATCH / 128, KSPLIT), 256, 0, stream>>>(
      x0, x1, x2, W1, h1part);
  k_bias_relu1<<<(BATCH * H1) / 256, 256, 0, stream>>>(h1part, b1, h1h);
  k_gemm2<<<dim3(H2 / 64, BATCH / 128), 256, 0, stream>>>(h1h, W2, b2, h2h);
  k_head<<<(BATCH * NCLS + 255) / 256, 256, 0, stream>>>(h2h, W3, b3, out);
}
